// ConditionInjectionBlock_45999099740391
// MI455X (gfx1250) — matrix operand order repaired and checked by reading, not compiled, not run on hardware
//
#include <hip/hip_runtime.h>
#include <hip/hip_bf16.h>

// ---------------- problem constants ----------------
#define DIMC 256
#define HEADS 8
#define HEAD_DIM 32
#define WIN 64          // window area (8x8)
#define NQ 128          // N1 = M * WIN
#define QSCALE 0.17677669529663687f   // 32^-0.5

// ---------------- workspace layout (bytes) ----------------
// W1 : kv_w   f16 [512][256]
// W2 : proj_w f16 [256][256]
// QF : q      f16 [128][256]  (pre-scaled by QSCALE)
// BG : bias   f32 [8][64][64] (rpb gathered by REL_IDX)
#define OFF_W1 0
#define OFF_W2 (OFF_W1 + 512*256*2)
#define OFF_QF (OFF_W2 + 256*256*2)
#define OFF_BG (OFF_QF + 128*256*2)
#define WS_BYTES (OFF_BG + 8*64*64*4)

// ---------------- LDS layout (halves / bytes) ----------------
// sMat : [128][264] f16  — rows 0..63 = window (phase 1), all rows = attn out (phase 3)
// sK   : [64][264]  f16  — keys, [key][h*32+d]
// sVT  : [32][520]  f16  — values transposed, [d][h*64+key]
// sP   : [8][16][72] f16 — per-wave softmax probabilities scratch
#define LDM 264
#define LDK 264
#define LDV 520
#define LDP 72
#define SOFF_K  (128*LDM*2)            // 67584
#define SOFF_VT (SOFF_K + 64*LDK*2)    // 101376
#define SOFF_P  (SOFF_VT + 32*LDV*2)   // 134656
#define SMEM_BYTES (SOFF_P + 8*16*LDP*2)  // 153088

typedef _Float16 v16h __attribute__((ext_vector_type(16)));
typedef _Float16 v8h  __attribute__((ext_vector_type(8)));
typedef float    v8f  __attribute__((ext_vector_type(8)));

static __device__ __forceinline__ v16h cat8(v8h lo, v8h hp) {
    return __builtin_shufflevector(lo, hp, 0,1,2,3,4,5,6,7,8,9,10,11,12,13,14,15);
}

// A fragment: source row-major [M][K], p points at (row m0, col k0), ld in halves.
// lane m = l&15 ; k(e) = ((e>>3)<<4) + 8*(l>>4) + (e&7)  -> two contiguous b128 loads
static __device__ __forceinline__ v16h load_a16(const _Float16* p, int ld, int lane) {
    const _Float16* r0 = p + (lane & 15) * ld + (lane >> 4) * 8;
    return cat8(*(const v8h*)r0, *(const v8h*)(r0 + 16));
}

// B fragment: source N-major [N][K] (i.e. D[m][n] = sum_k A[m][k]*Bsrc[n][k]).
// lane n = l&15 ; k(e) = ((e>>3)<<4) + 8*(l>>4) + (e&7)  -> two contiguous b128 loads
static __device__ __forceinline__ v16h load_b16(const _Float16* p, int ld, int lane) {
    const _Float16* r0 = p + (lane & 15) * ld + (lane >> 4) * 8;
    return cat8(*(const v8h*)r0, *(const v8h*)(r0 + 16));
}

static __device__ __forceinline__ v8f wmma_f16(v16h a, v16h b, v8f c) {
    return __builtin_amdgcn_wmma_f32_16x16x32_f16(false, a, false, b, (short)0, c,
                                                  false, false);
}

// ---------------- prep: weight/f16 conversion, bias gather ----------------
__global__ void cib_prep_kernel(const float* __restrict__ embeds,
                                const float* __restrict__ rpb,
                                const float* __restrict__ kv_w,
                                const float* __restrict__ proj_w,
                                char* __restrict__ ws) {
    _Float16* W1 = (_Float16*)(ws + OFF_W1);
    _Float16* W2 = (_Float16*)(ws + OFF_W2);
    _Float16* QF = (_Float16*)(ws + OFF_QF);
    float*    BG = (float*)(ws + OFF_BG);
    const int tid = blockIdx.x * blockDim.x + threadIdx.x;
    const int nth = gridDim.x * blockDim.x;
    for (int i = tid; i < 512 * 256; i += nth) W1[i] = (_Float16)kv_w[i];
    for (int i = tid; i < 256 * 256; i += nth) W2[i] = (_Float16)proj_w[i];
    for (int i = tid; i < 128 * 256; i += nth) QF[i] = (_Float16)(embeds[i] * QSCALE);
    for (int i = tid; i < 8 * 64 * 64; i += nth) {
        const int h  = i >> 12;
        const int qi = (i >> 6) & 63;
        const int kj = i & 63;
        const int dh = (qi >> 3) - (kj >> 3) + 7;
        const int dw = (qi & 7) - (kj & 7) + 7;
        BG[i] = rpb[(dh * 15 + dw) * HEADS + h];
    }
}

// ---------------- fused per-window attention block ----------------
__global__ __launch_bounds__(256, 1)
void cib_attn_kernel(const float* __restrict__ x,
                     const float* __restrict__ kv_b,
                     const float* __restrict__ proj_b,
                     const char* __restrict__ ws,
                     float* __restrict__ out) {
    extern __shared__ char smem[];
    _Float16* sMat = (_Float16*)(smem);
    _Float16* sK   = (_Float16*)(smem + SOFF_K);
    _Float16* sVT  = (_Float16*)(smem + SOFF_VT);
    _Float16* sP   = (_Float16*)(smem + SOFF_P);

    const _Float16* W1 = (const _Float16*)(ws + OFF_W1);
    const _Float16* W2 = (const _Float16*)(ws + OFF_W2);
    const _Float16* QF = (const _Float16*)(ws + OFF_QF);
    const float*    BG = (const float*)(ws + OFF_BG);

    const int tid  = threadIdx.x;
    const int lane = tid & 31;
    const int w    = tid >> 5;     // wave 0..7
    const int l15  = lane & 15;
    const int hi   = lane >> 4;

    const int b_  = blockIdx.x;    // window id 0..2047
    const int img = b_ >> 10;
    const int wid = b_ & 1023;
    const int wh  = wid >> 5, wwx = wid & 31;

    // ---- phase 0: stage the 64x256 window into LDS as f16 ----
    {
        const float* xb = x + (size_t)img * 256 * 256 * DIMC;
        #pragma unroll
        for (int it = 0; it < 16; ++it) {
            const int f   = it * 256 + tid;    // float4 id 0..4095
            const int row = f >> 6;            // window row 0..63
            const int c4  = f & 63;
            const int py  = wh * 8 + (row >> 3);
            const int px  = wwx * 8 + (row & 7);
            const float4 v4 = *(const float4*)(xb + ((size_t)py * 256 + px) * DIMC + c4 * 4);
            _Float16* d = sMat + row * LDM + c4 * 4;
            d[0] = (_Float16)v4.x; d[1] = (_Float16)v4.y;
            d[2] = (_Float16)v4.z; d[3] = (_Float16)v4.w;
        }
    }
    __syncthreads();

    // ---- phase 1: kv = win @ kv_w^T + kv_b  (64x512, K=256) ----
    for (int t = w; t < 128; t += 8) {
        const int mt = t >> 5;                 // 0..3
        const int nt = t & 31;                 // 0..31
        __builtin_prefetch(W1 + (size_t)(nt * 16) * 256, 0, 1);
        v8f acc = {};
        #pragma unroll
        for (int kk = 0; kk < 8; ++kk) {
            v16h a = load_a16(sMat + mt * 16 * LDM + kk * 32, LDM, lane);
            v16h b = load_b16(W1 + (size_t)(nt * 16) * 256 + kk * 32, 256, lane);
            acc = wmma_f16(a, b, acc);
        }
        const int n = nt * 16 + l15;           // 0..511
        const float bb = kv_b[n];
        #pragma unroll
        for (int r = 0; r < 8; ++r) {
            const int m = mt * 16 + r + hi * 8;      // key 0..63
            const float vv = acc[r] + bb;
            if (n < 256) {
                sK[m * LDK + n] = (_Float16)vv;
            } else {
                const int c = n - 256;
                sVT[(c & 31) * LDV + (c >> 5) * 64 + m] = (_Float16)vv;
            }
        }
    }
    __syncthreads();

    // ---- phase 2: per-head attention (wave w owns head w) ----
    {
        const int h = w;
        _Float16* probs = sP + w * 16 * LDP;
        const float* bg = BG + h * 4096;
        for (int mt = 0; mt < 8; ++mt) {
            const int m0 = mt * 16;
            const v16h aq = load_a16(QF + (size_t)m0 * 256 + h * 32, 256, lane);
            v8f logit[4];
            #pragma unroll
            for (int nt = 0; nt < 4; ++nt) {
                v8f c = {};
                v16h b = load_b16(sK + nt * 16 * LDK + h * 32, LDK, lane);
                logit[nt] = wmma_f16(aq, b, c);
            }
            // bias + in-register softmax over 64 keys (4 vals/lane x 16 lanes)
            float inv[8];
            #pragma unroll
            for (int r = 0; r < 8; ++r) {
                const int qrow = (m0 + r + hi * 8) & 63;
                float mx = -1e30f;
                #pragma unroll
                for (int nt = 0; nt < 4; ++nt) {
                    const float vv = logit[nt][r] + bg[qrow * 64 + nt * 16 + l15];
                    logit[nt][r] = vv;
                    mx = fmaxf(mx, vv);
                }
                for (int s = 8; s >= 1; s >>= 1) mx = fmaxf(mx, __shfl_xor(mx, s, 32));
                float sum = 0.f;
                #pragma unroll
                for (int nt = 0; nt < 4; ++nt) {
                    const float e = __expf(logit[nt][r] - mx);
                    logit[nt][r] = e;
                    sum += e;
                }
                for (int s = 8; s >= 1; s >>= 1) sum += __shfl_xor(sum, s, 32);
                inv[r] = 1.f / sum;
            }
            #pragma unroll
            for (int nt = 0; nt < 4; ++nt)
                #pragma unroll
                for (int r = 0; r < 8; ++r)
                    probs[(r + hi * 8) * LDP + nt * 16 + l15] =
                        (_Float16)(logit[nt][r] * inv[r]);
            // PV: out[16x32] = probs[16x64] @ V[64x32]
            #pragma unroll
            for (int dt = 0; dt < 2; ++dt) {
                v8f o = {};
                #pragma unroll
                for (int kk = 0; kk < 2; ++kk) {
                    v16h a = load_a16(probs + kk * 32, LDP, lane);
                    v16h b = load_b16(sVT + dt * 16 * LDV + h * 64 + kk * 32, LDV, lane);
                    o = wmma_f16(a, b, o);
                }
                #pragma unroll
                for (int r = 0; r < 8; ++r)
                    sMat[(m0 + r + hi * 8) * LDM + h * 32 + dt * 16 + l15] = (_Float16)o[r];
            }
        }
    }
    __syncthreads();

    // ---- phase 3: out = attn_out @ proj_w^T + proj_b  (128x256, K=256) ----
    for (int t = w; t < 128; t += 8) {
        const int mt = t >> 4;                 // 0..7
        const int nt = t & 15;                 // 0..15
        __builtin_prefetch(W2 + (size_t)(nt * 16) * 256, 0, 1);
        v8f acc = {};
        #pragma unroll
        for (int kk = 0; kk < 8; ++kk) {
            v16h a = load_a16(sMat + mt * 16 * LDM + kk * 32, LDM, lane);
            v16h b = load_b16(W2 + (size_t)(nt * 16) * 256 + kk * 32, 256, lane);
            acc = wmma_f16(a, b, acc);
        }
        const int n = nt * 16 + l15;
        const float pb = proj_b[n];
        float* ob = out + ((size_t)b_ * NQ + mt * 16 + hi * 8) * DIMC + n;
        #pragma unroll
        for (int r = 0; r < 8; ++r)
            ob[(size_t)r * DIMC] = acc[r] + pb;
    }
}

extern "C" void kernel_launch(void* const* d_in, const int* in_sizes, int n_in,
                              void* d_out, int out_size, void* d_ws, size_t ws_size,
                              hipStream_t stream) {
    const float* embeds = (const float*)d_in[0];
    const float* x      = (const float*)d_in[1];
    const float* rpb    = (const float*)d_in[2];
    const float* kv_w   = (const float*)d_in[3];
    const float* kv_b   = (const float*)d_in[4];
    const float* proj_w = (const float*)d_in[5];
    const float* proj_b = (const float*)d_in[6];
    float* out = (float*)d_out;
    char*  ws  = (char*)d_ws;
    (void)in_sizes; (void)n_in; (void)out_size; (void)ws_size;

    cib_prep_kernel<<<256, 256, 0, stream>>>(embeds, rpb, kv_w, proj_w, ws);
    cib_attn_kernel<<<2048, 256, SMEM_BYTES, stream>>>(x, kv_b, proj_b, ws, out);
}
